// SelfAttention_fuse_38259568673590
// MI455X (gfx1250) — matrix operand order repaired and checked by reading, not compiled, not run on hardware
//
#include <hip/hip_runtime.h>

#define B_    2
#define C_    256
#define NSP   32768      // 32*32*32
#define HS    64         // head size == group size

typedef __attribute__((ext_vector_type(16))) __bf16 bf16x16;
typedef __attribute__((ext_vector_type(8)))  float  f32x8;

union BfFrag {
  unsigned short u[16];
  uint4          q[2];
  bf16x16        v;
};

__device__ __forceinline__ unsigned short f2bf(float x){
  unsigned u = __float_as_uint(x);
  u += 0x7FFFu + ((u >> 16) & 1u);          // round-to-nearest-even
  return (unsigned short)(u >> 16);
}
__device__ __forceinline__ float bf2f(unsigned short u){
  return __uint_as_float(((unsigned)u) << 16);
}

// ---------------- init: zero sums / rowstats / context -------------------
__global__ void k_init(float* stats){
  int i = blockIdx.x * 256 + threadIdx.x;
  if (i < 33856) stats[i] = 0.f;
}

// ---------------- groupnorm stats: sum & sumsq per (tensor,b,g) ----------
__global__ void k_gn_stats(const float* __restrict__ q, const float* __restrict__ k,
                           const float* __restrict__ v, float* __restrict__ sums){
  const int CHUNK = 65536;                       // 2M elems / 32 chunks
  int grp   = blockIdx.x >> 5;                   // 0..23 : t*8 + b*4 + g
  int chunk = blockIdx.x & 31;
  int t  = grp >> 3;
  int bg = grp & 7;                              // b*4+g ; group is contiguous in mem
  const float* src = (t == 0) ? q : ((t == 1) ? k : v);
  size_t base = (size_t)bg * ((size_t)HS * NSP) + (size_t)chunk * CHUNK;
  float s = 0.f, ss = 0.f;
  for (int i = threadIdx.x; i < CHUNK; i += 256){
    float x = src[base + i];
    s += x; ss += x * x;
  }
  __shared__ float red0[256], red1[256];
  red0[threadIdx.x] = s; red1[threadIdx.x] = ss;
  __syncthreads();
  for (int off = 128; off > 0; off >>= 1){
    if (threadIdx.x < off){
      red0[threadIdx.x] += red0[threadIdx.x + off];
      red1[threadIdx.x] += red1[threadIdx.x + off];
    }
    __syncthreads();
  }
  if (threadIdx.x == 0){
    atomicAdd(&sums[grp * 2 + 0], red0[0]);
    atomicAdd(&sums[grp * 2 + 1], red1[0]);
  }
}

__global__ void k_gn_fin(float* sums){
  int i = threadIdx.x;
  if (i >= 24) return;
  float cnt = (float)((size_t)HS * NSP);
  float mu  = sums[i * 2] / cnt;
  float var = sums[i * 2 + 1] / cnt - mu * mu;
  sums[i * 2]     = mu;
  sums[i * 2 + 1] = rsqrtf(var + 1e-5f);
}

// ---------------- softmax row stats over n for nk ------------------------
__global__ void k_softmax_stats(const float* __restrict__ k, const float* __restrict__ gnw,
                                const float* __restrict__ gnb, const float* __restrict__ stats,
                                float* __restrict__ rowmax, float* __restrict__ rowrcp){
  int row = blockIdx.x;                 // b*256 + ch , 512 rows
  int b = row >> 8, ch = row & 255, g = ch >> 6;
  int sidx = (8 + b * 4 + g) * 2;       // tensor k
  float mu = stats[sidx], rs = stats[sidx + 1];
  float sc = rs * gnw[ch];
  float sh = gnb[ch] - mu * sc;
  const float* src = k + (size_t)row * NSP;
  __shared__ float red[256];
  float m = -3.4e38f;
  for (int i = threadIdx.x; i < NSP; i += 256)
    m = fmaxf(m, src[i] * sc + sh);
  red[threadIdx.x] = m; __syncthreads();
  for (int off = 128; off > 0; off >>= 1){
    if (threadIdx.x < off) red[threadIdx.x] = fmaxf(red[threadIdx.x], red[threadIdx.x + off]);
    __syncthreads();
  }
  m = red[0]; __syncthreads();
  float s = 0.f;
  for (int i = threadIdx.x; i < NSP; i += 256)
    s += __expf(src[i] * sc + sh - m);
  red[threadIdx.x] = s; __syncthreads();
  for (int off = 128; off > 0; off >>= 1){
    if (threadIdx.x < off) red[threadIdx.x] += red[threadIdx.x + off];
    __syncthreads();
  }
  if (threadIdx.x == 0){ rowmax[row] = m; rowrcp[row] = 1.f / red[0]; }
}

// ------- pack nq = groupnorm(q) to bf16, TRANSPOSED to [b][n][ci] --------
// 64ch x 64pos tile through LDS: coalesced reads along pos, coalesced
// 128B writes along ci.  nqT feeds the proj WMMA B operand with b128 loads.
__global__ void k_nq_packT(const float* __restrict__ q, const float* __restrict__ gnw,
                           const float* __restrict__ gnb, const float* __restrict__ stats,
                           unsigned short* __restrict__ nqT){
  __shared__ unsigned short tile[64][68];        // pad to dodge bank conflicts
  int blk  = blockIdx.x;                         // 2 * 4 * 512 = 4096 tiles
  int b    = blk >> 11;
  int chT  = (blk >> 9) & 3;
  int posT = blk & 511;
  int ch0  = chT * 64;
  int pos0 = posT * 64;
  int t = threadIdx.x;

  // read + normalize: e = iter*256 + t, pos fastest (coalesced)
#pragma unroll
  for (int iter = 0; iter < 16; ++iter){
    int e = iter * 256 + t;
    int cl = e >> 6;                 // 0..63 channel-local
    int pl = e & 63;                 // pos-local
    int ch = ch0 + cl;
    int g  = ch >> 6;
    int sidx = (0 + b * 4 + g) * 2;  // tensor q
    float mu = stats[sidx], rs = stats[sidx + 1];
    float x = q[((size_t)(b * C_) + ch) * NSP + pos0 + pl];
    tile[cl][pl] = f2bf((x - mu) * rs * gnw[ch] + gnb[ch]);
  }
  __syncthreads();
  // write transposed: e = iter*256 + t, ci fastest (coalesced 128B rows)
#pragma unroll
  for (int iter = 0; iter < 16; ++iter){
    int e = iter * 256 + t;
    int ci = e & 63;
    int pl = e >> 6;
    nqT[((size_t)(b * NSP) + pos0 + pl) * C_ + ch0 + ci] = tile[ci][pl];
  }
}

// ---------------- context[b,h,d,e] = sum_n kh[d,n]*nv[e,n]  (WMMA) -------
__global__ void k_context(const float* __restrict__ kk, const float* __restrict__ vv,
                          const float* __restrict__ gnw, const float* __restrict__ gnb,
                          const float* __restrict__ stats, const float* __restrict__ rowmax,
                          const float* __restrict__ rowrcp, float* __restrict__ ctx){
  int bh   = blockIdx.x;                  // 0..7 : b*4+h
  int kc   = blockIdx.y;                  // 0..31 : K chunk of 1024
  int wave = threadIdx.x >> 5;            // 0..15
  int lane = threadIdx.x & 31;
  int d0 = (wave >> 2) * 16;
  int e0 = (wave &  3) * 16;
  int b = bh >> 2, h = bh & 3, g = h;     // groups align with heads
  int lg = lane >> 4, lm = lane & 15;

  int chA = h * HS + d0 + lm;             // A row channel (kh)
  int chB = h * HS + e0 + lm;             // B row channel (nv)
  int rowA = b * C_ + chA;
  // gn affine for k (tensor 1) and v (tensor 2)
  int ksi = (8 + b * 4 + g) * 2, vsi = (16 + b * 4 + g) * 2;
  float scA = stats[ksi + 1] * gnw[chA];
  float shA = gnb[chA] - stats[ksi] * scA;
  float scB = stats[vsi + 1] * gnw[chB];
  float shB = gnb[chB] - stats[vsi] * scB;
  float aM = rowmax[rowA], aR = rowrcp[rowA];

  const float* arow = kk + (size_t)rowA * NSP;
  const float* brow = vv + ((size_t)(b * C_) + chB) * NSP;

  f32x8 acc = {0.f,0.f,0.f,0.f,0.f,0.f,0.f,0.f};
  for (int it = 0; it < 32; ++it){
    int n0 = kc * 1024 + it * 32;
    BfFrag A, Bf;
    float af[16], bfv[16];
    *(float4*)(af  + 0)  = *(const float4*)(arow + n0 + 8 * lg);
    *(float4*)(af  + 4)  = *(const float4*)(arow + n0 + 8 * lg + 4);
    *(float4*)(af  + 8)  = *(const float4*)(arow + n0 + 16 + 8 * lg);
    *(float4*)(af  + 12) = *(const float4*)(arow + n0 + 16 + 8 * lg + 4);
    *(float4*)(bfv + 0)  = *(const float4*)(brow + n0 + 8 * lg);
    *(float4*)(bfv + 4)  = *(const float4*)(brow + n0 + 8 * lg + 4);
    *(float4*)(bfv + 8)  = *(const float4*)(brow + n0 + 16 + 8 * lg);
    *(float4*)(bfv + 12) = *(const float4*)(brow + n0 + 16 + 8 * lg + 4);
#pragma unroll
    for (int j = 0; j < 16; ++j){
      A.u[j]  = f2bf(__expf(af[j] * scA + shA - aM) * aR);   // softmaxed nk
      Bf.u[j] = f2bf(bfv[j] * scB + shB);                    // nv
    }
    acc = __builtin_amdgcn_wmma_f32_16x16x32_bf16(false, A.v, false, Bf.v,
                                                  (short)0, acc, false, false);
  }
#pragma unroll
  for (int r = 0; r < 8; ++r){
    int M = r + 8 * lg, N = lm;
    atomicAdd(&ctx[(((size_t)bh * HS) + d0 + M) * HS + e0 + N], acc[r]);
  }
}

// ---------------- Weff[b,co,hd] = sum_e out_w[co,h*64+e] * ctx[b,h,d,e] --
__global__ void k_weff(const float* __restrict__ outw, const float* __restrict__ ctx,
                       unsigned short* __restrict__ weff){
  int t = blockIdx.x * 256 + threadIdx.x;    // < 131072
  int b = t >> 16, co = (t >> 8) & 255, hd = t & 255;
  int h = hd >> 6, d = hd & 63;
  const float* wrow = outw + (size_t)co * C_ + h * HS;
  const float* crow = ctx + (((size_t)(b * 4 + h) * HS) + d) * HS;
  float s = 0.f;
  for (int e = 0; e < HS; ++e) s += wrow[e] * crow[e];
  weff[t] = f2bf(s);
}

// ---------------- proj + residual: y = Weff @ nq + out_b + q  (WMMA) -----
// A (Weff) and B (nqT, n-major) both load as aligned b128 pairs.
__global__ void k_proj(const unsigned short* __restrict__ weff,
                       const unsigned short* __restrict__ nqT,
                       const float* __restrict__ qres, const float* __restrict__ outb,
                       unsigned short* __restrict__ y){
  int b    = blockIdx.z;
  int m0   = blockIdx.y * 16;                 // co tile
  int wave = threadIdx.x >> 5;                // 0..7
  int lane = threadIdx.x & 31;
  int n0   = (blockIdx.x * 8 + wave) * 16;    // n tile
  int lg = lane >> 4, lm = lane & 15;

  const unsigned short* arow = weff + ((size_t)b * C_ + m0 + lm) * C_;
  const unsigned short* brow = nqT + ((size_t)(b * NSP) + n0 + lm) * C_;
  f32x8 acc = {0.f,0.f,0.f,0.f,0.f,0.f,0.f,0.f};
#pragma unroll
  for (int k0 = 0; k0 < C_; k0 += 32){
    BfFrag A, Bf;
    A.q[0]  = *(const uint4*)(arow + k0 + 8 * lg);
    A.q[1]  = *(const uint4*)(arow + k0 + 16 + 8 * lg);
    Bf.q[0] = *(const uint4*)(brow + k0 + 8 * lg);
    Bf.q[1] = *(const uint4*)(brow + k0 + 16 + 8 * lg);
    acc = __builtin_amdgcn_wmma_f32_16x16x32_bf16(false, A.v, false, Bf.v,
                                                  (short)0, acc, false, false);
  }
#pragma unroll
  for (int r = 0; r < 8; ++r){
    int co = m0 + r + 8 * lg;
    size_t o = ((size_t)b * C_ + co) * NSP + n0 + lm;
    y[o] = f2bf(acc[r] + outb[co] + qres[o]);
  }
}

// ---------------- 3x3x3 conv, 256 -> 3 channels --------------------------
__global__ void k_dgconv(const unsigned short* __restrict__ y, const float* __restrict__ dgw,
                         const float* __restrict__ dgb, float* __restrict__ dgo){
  int t = blockIdx.x * 256 + threadIdx.x;    // < 2*3*32768
  int pos = t & 32767;
  int o = (t >> 15) % 3;
  int b = t / (3 * 32768);
  int wq = pos & 31, hq = (pos >> 5) & 31, dq = pos >> 10;
  float acc = dgb[o];
  for (int kd = 0; kd < 3; ++kd){
    int zd = dq + kd - 1; if ((unsigned)zd >= 32u) continue;
    for (int kh = 0; kh < 3; ++kh){
      int zh = hq + kh - 1; if ((unsigned)zh >= 32u) continue;
      for (int kw = 0; kw < 3; ++kw){
        int zw = wq + kw - 1; if ((unsigned)zw >= 32u) continue;
        const unsigned short* yp = y + (size_t)b * C_ * NSP + (size_t)(zd * 1024 + zh * 32 + zw);
        const float* wp = dgw + (size_t)o * C_ * 27 + (kd * 9 + kh * 3 + kw);
        float s = 0.f;
        for (int ci = 0; ci < C_; ++ci)
          s += wp[(size_t)ci * 27] * bf2f(yp[(size_t)ci * NSP]);
        acc += s;
      }
    }
  }
  dgo[t] = acc;
}

// ---------------- fused nearest-x2 upsample + 3x3x3 conv 3 -> 3 ----------
__global__ void k_finalconv(const float* __restrict__ dgo, const float* __restrict__ nlw,
                            const float* __restrict__ nlb, float* __restrict__ out){
  int t = blockIdx.x * 256 + threadIdx.x;    // < 2*3*262144
  int pos = t & 262143;
  int o = (t >> 18) % 3;
  int b = t / (3 * 262144);
  int wq = pos & 63, hq = (pos >> 6) & 63, dq = pos >> 12;
  float acc = nlb[o];
  for (int ci = 0; ci < 3; ++ci){
    const float* src = dgo + (size_t)(b * 3 + ci) * 32768;
    const float* wp  = nlw + (size_t)(o * 3 + ci) * 27;
    for (int kd = 0; kd < 3; ++kd){
      int zd = dq + kd - 1; if ((unsigned)zd >= 64u) continue;
      for (int kh = 0; kh < 3; ++kh){
        int zh = hq + kh - 1; if ((unsigned)zh >= 64u) continue;
        for (int kw = 0; kw < 3; ++kw){
          int zw = wq + kw - 1; if ((unsigned)zw >= 64u) continue;
          acc += wp[kd * 9 + kh * 3 + kw] *
                 src[(zd >> 1) * 1024 + (zh >> 1) * 32 + (zw >> 1)];
        }
      }
    }
  }
  out[t] = acc;
}

extern "C" void kernel_launch(void* const* d_in, const int* in_sizes, int n_in,
                              void* d_out, int out_size, void* d_ws, size_t ws_size,
                              hipStream_t stream){
  (void)in_sizes; (void)n_in; (void)out_size; (void)ws_size;
  const float* q    = (const float*)d_in[0];
  const float* k    = (const float*)d_in[1];
  const float* v    = (const float*)d_in[2];
  const float* gnw  = (const float*)d_in[3];
  const float* gnb  = (const float*)d_in[4];
  const float* outw = (const float*)d_in[5];
  const float* outb = (const float*)d_in[6];
  const float* dgw  = (const float*)d_in[7];
  const float* dgb  = (const float*)d_in[8];
  const float* nlw  = (const float*)d_in[9];
  const float* nlb  = (const float*)d_in[10];
  // d_in[11] = size (== 64), folded into k_finalconv indexing

  char* ws = (char*)d_ws;
  unsigned short* nqT = (unsigned short*)ws;                        // 33,554,432 B  [b][n][ci]
  unsigned short* yb  = (unsigned short*)(ws + 33554432);           // 33,554,432 B  [b][ch][n]
  float* stats  = (float*)(ws + 67108864);
  float* sums   = stats;                                            // [24][2]
  float* rowmax = stats + 64;                                       // [512]
  float* rowrcp = stats + 576;                                      // [512]
  float* ctx    = stats + 1088;                                     // [2*4*64*64]
  unsigned short* weff = (unsigned short*)(stats + 1088 + 32768);   // [2*256*256]
  float* dgo    = (float*)((char*)weff + 262144);                   // [2*3*32768]
  float* out    = (float*)d_out;

  k_init<<<133, 256, 0, stream>>>(stats);
  k_gn_stats<<<768, 256, 0, stream>>>(q, k, v, sums);
  k_gn_fin<<<1, 32, 0, stream>>>(sums);
  k_softmax_stats<<<512, 256, 0, stream>>>(k, gnw, gnb, sums, rowmax, rowrcp);
  k_nq_packT<<<4096, 256, 0, stream>>>(q, gnw, gnb, sums, nqT);
  dim3 gctx(8, 32);
  k_context<<<gctx, 512, 0, stream>>>(k, v, gnw, gnb, sums, rowmax, rowrcp, ctx);
  k_weff<<<512, 256, 0, stream>>>(outw, ctx, weff);
  dim3 gproj(256, 16, 2);
  k_proj<<<gproj, 256, 0, stream>>>(weff, nqT, q, outb, yb);
  k_dgconv<<<768, 256, 0, stream>>>(yb, dgw, dgb, dgo);
  k_finalconv<<<6144, 256, 0, stream>>>(dgo, nlw, nlb, out);
}
